// TemporalGraphTransformerLayer_28381143892400
// MI455X (gfx1250) — compile-verified, matrix operand order checked by reading, not run on hardware
//
#include <hip/hip_runtime.h>
#include <hip/hip_bf16.h>

#define NNODE 50000
#define NEDGE 600000
#define HID 128
#define NH 4
#define HD 32
#define FFDIM 512

typedef __attribute__((ext_vector_type(16))) _Float16 v16h;
typedef __attribute__((ext_vector_type(8)))  float    v8f;

// ---------------- WMMA tile helpers (16x16 C tile, f16 A/B, f32 acc) ----------------
// A fragment: 16x32 f16. Lanes 0-15 hold row M=lane, K {k0..k0+7, k0+16..k0+23};
// lanes 16-31 hold row M=lane-16, K {k0+8..k0+15, k0+24..k0+31}. (ISA 7.12.2)
__device__ __forceinline__ v16h ld_a(const _Float16* A, int lda, int k0, int lane) {
  int m  = lane & 15;
  int kh = (lane >> 4) & 1;
  const _Float16* p = A + m * lda + k0 + kh * 8;
  v16h r;
#pragma unroll
  for (int i = 0; i < 8; ++i) r[i] = p[i];
#pragma unroll
  for (int i = 0; i < 8; ++i) r[8 + i] = p[16 + i];
  return r;
}

// Packed-B fragment: weights pre-packed so each (nt, kt, lane) fragment is a
// contiguous 32B v16h -> lowers to 2x global_load_b128 (L2-resident).
__device__ __forceinline__ v16h ld_bp(const _Float16* Wp, int KT, int kt, int nt, int lane) {
  return ((const v16h*)Wp)[(nt * KT + kt) * 32 + lane];
}

__device__ __forceinline__ v8f wmma_step(v16h a, v16h b, v8f c) {
  return __builtin_amdgcn_wmma_f32_16x16x32_f16(false, a, false, b, (short)0, c, false, false);
}

template <int K>
__device__ __forceinline__ v8f tile_gemm(const _Float16* A, int lda, const _Float16* Wp,
                                         int nt, int lane) {
  constexpr int KT = K / 32;
  v8f acc = {};
#pragma unroll
  for (int kt = 0; kt < KT; ++kt)
    acc = wmma_step(ld_a(A, lda, kt * 32, lane), ld_bp(Wp, KT, kt, nt, lane), acc);
  return acc;
}

// C layout: element j of lane L is (M = j + 8*(L>=16), N = L%16). (ISA 7.12.2)

// ---------------- math helpers ----------------
__device__ __forceinline__ float gelu_f(float v) { return 0.5f * v * (1.f + erff(v * 0.70710678118654752f)); }
__device__ __forceinline__ float sigm_f(float v) { return 1.f / (1.f + expf(-v)); }
__device__ __forceinline__ unsigned enc_f(float f) {
  unsigned u = __float_as_uint(f);
  return (u & 0x80000000u) ? ~u : (u | 0x80000000u);
}
__device__ __forceinline__ float dec_f(unsigned u) {
  return (u & 0x80000000u) ? __uint_as_float(u & 0x7FFFFFFFu) : __uint_as_float(~u);
}

// row-wise LayerNorm of Y[16][128] then store
__device__ __forceinline__ void ln_store(float (*Y)[HID], float (*red1)[8], float (*red2)[8],
                                         float* mu, float* rs, int t, float* out) {
  {
    int r = t >> 3, seg = t & 7;
    float s1 = 0.f, s2 = 0.f;
#pragma unroll
    for (int i = 0; i < 16; ++i) { float v = Y[r][seg * 16 + i]; s1 += v; s2 += v * v; }
    red1[r][seg] = s1; red2[r][seg] = s2;
  }
  __syncthreads();
  if (t < 16) {
    float s1 = 0.f, s2 = 0.f;
#pragma unroll
    for (int i = 0; i < 8; ++i) { s1 += red1[t][i]; s2 += red2[t][i]; }
    float m = s1 * (1.f / HID);
    float v = s2 * (1.f / HID) - m * m;
    mu[t] = m; rs[t] = rsqrtf(v + 1e-5f);
  }
  __syncthreads();
  for (int r = 0; r < 16; ++r)
    out[(size_t)r * HID + t] = (Y[r][t] - mu[r]) * rs[r];
}

// ---------------- kernels ----------------
// pack f32 weight W[K][N] into WMMA B-fragment-major f16: frag(nt,kt) x lane x 16h
__global__ __launch_bounds__(256) void k_pack(const float* __restrict__ W, int K, int N,
                                              _Float16* __restrict__ out) {
  int tid = blockIdx.x * blockDim.x + threadIdx.x;
  int KT = K >> 5;
  int total = (N >> 4) * KT * 32;
  if (tid >= total) return;
  int lane = tid & 31;
  int frag = tid >> 5;
  int kt = frag % KT;
  int nt = frag / KT;
  int n  = nt * 16 + (lane & 15);
  int kb = kt * 32 + ((lane >> 4) & 1) * 8;
  v16h r;
#pragma unroll
  for (int i = 0; i < 8; ++i) r[i]     = (_Float16)W[(kb + i) * N + n];
#pragma unroll
  for (int i = 0; i < 8; ++i) r[8 + i] = (_Float16)W[(kb + 16 + i) * N + n];
  ((v16h*)out)[tid] = r;
}

__global__ __launch_bounds__(256) void k_init(unsigned* smaxu, float* ssum, float* agg, float* agg2) {
  int i = blockIdx.x * blockDim.x + threadIdx.x;
  if (i < NNODE * HID) { agg[i] = 0.f; agg2[i] = 0.f; }
  if (i < NNODE * NH)  { smaxu[i] = 0u; ssum[i] = 0.f; }
}

// scores[e][h] = (q.k)/sqrt(HD) + tanh(tp*wt+bt); q from x[dst], k from x[src]+ge
__global__ __launch_bounds__(128) void k_scores(
    const float* __restrict__ x, const int* __restrict__ ei, const float* __restrict__ ge,
    const float* __restrict__ tp,
    const _Float16* __restrict__ pWq, const float* __restrict__ bq,
    const _Float16* __restrict__ pWk, const float* __restrict__ bk,
    const float* __restrict__ wt, const float* __restrict__ bt,
    float* __restrict__ sc) {
  __shared__ _Float16 Axi[16][HID];
  __shared__ _Float16 Akj[16][HID];
  const int e0 = blockIdx.x * 16;
  const int t = threadIdx.x, lane = t & 31, w = t >> 5;
  const int* src = ei;
  const int* dst = ei + NEDGE;
  for (int r = 0; r < 16; ++r) {
    int e = e0 + r;
    int s = src[e], d = dst[e];
    Axi[r][t] = (_Float16)x[(size_t)d * HID + t];
    Akj[r][t] = (_Float16)(x[(size_t)s * HID + t] + ge[(size_t)e * HID + t]);
  }
  if (e0 + 16 < NEDGE) __builtin_prefetch(ge + (size_t)(e0 + 16) * HID, 0, 1);
  __syncthreads();

  const int n0 = w * HD;  // head w owns cols [32w, 32w+32)
  v8f q0 = tile_gemm<HID>(&Axi[0][0], HID, pWq, 2 * w,     lane);
  v8f q1 = tile_gemm<HID>(&Axi[0][0], HID, pWq, 2 * w + 1, lane);
  v8f k0 = tile_gemm<HID>(&Akj[0][0], HID, pWk, 2 * w,     lane);
  v8f k1 = tile_gemm<HID>(&Akj[0][0], HID, pWk, 2 * w + 1, lane);
  float bq0 = bq[n0 + (lane & 15)], bq1 = bq[n0 + 16 + (lane & 15)];
  float bk0 = bk[n0 + (lane & 15)], bk1 = bk[n0 + 16 + (lane & 15)];
  float p[8];
#pragma unroll
  for (int j = 0; j < 8; ++j)
    p[j] = (q0[j] + bq0) * (k0[j] + bk0) + (q1[j] + bq1) * (k1[j] + bk1);
#pragma unroll
  for (int j = 0; j < 8; ++j) {
#pragma unroll
    for (int m = 1; m < 16; m <<= 1) p[j] += __shfl_xor(p[j], m, 32);
  }
  if ((lane & 15) == 0) {
    int rbase = (lane >> 4) * 8;
    const float inv = 0.17677669529663687f;  // 1/sqrt(32)
#pragma unroll
    for (int j = 0; j < 8; ++j) {
      int e = e0 + rbase + j;
      sc[(size_t)e * NH + w] = p[j] * inv + tanhf(tp[e] * wt[w] + bt[w]);
    }
  }
}

__global__ __launch_bounds__(256) void k_smax(const int* __restrict__ ei,
                                              const float* __restrict__ sc,
                                              unsigned* __restrict__ smaxu) {
  int i = blockIdx.x * blockDim.x + threadIdx.x;
  if (i >= NEDGE * NH) return;
  int e = i >> 2, hh = i & 3;
  int d = ei[NEDGE + e];
  atomicMax(&smaxu[d * NH + hh], enc_f(sc[i]));
}

__global__ __launch_bounds__(256) void k_expsum(const int* __restrict__ ei,
                                                float* __restrict__ sc,
                                                const unsigned* __restrict__ smaxu,
                                                float* __restrict__ ssum) {
  int i = blockIdx.x * blockDim.x + threadIdx.x;
  if (i >= NEDGE * NH) return;
  int e = i >> 2, hh = i & 3;
  int d = ei[NEDGE + e];
  float ex = expf(sc[i] - dec_f(smaxu[d * NH + hh]));
  sc[i] = ex;
  atomicAdd(&ssum[d * NH + hh], ex);
}

// recompute v = (x[src]+ge)@Wv+bv, scatter attn*v into agg
__global__ __launch_bounds__(128) void k_aggv(
    const float* __restrict__ x, const int* __restrict__ ei, const float* __restrict__ ge,
    const _Float16* __restrict__ pWv, const float* __restrict__ bv,
    const float* __restrict__ sc, const float* __restrict__ ssum, float* __restrict__ agg) {
  __shared__ _Float16 Akj[16][HID];
  __shared__ float att[NH][16];
  __shared__ int dstS[16];
  const int e0 = blockIdx.x * 16;
  const int t = threadIdx.x, lane = t & 31, w = t >> 5;
  const int* src = ei;
  const int* dst = ei + NEDGE;
  for (int r = 0; r < 16; ++r) {
    int e = e0 + r;
    Akj[r][t] = (_Float16)(x[(size_t)src[e] * HID + t] + ge[(size_t)e * HID + t]);
  }
  if (t < 16) dstS[t] = dst[e0 + t];
  if (lane < 16) {
    int e = e0 + lane;
    att[w][lane] = sc[(size_t)e * NH + w] / ssum[dst[e] * NH + w];
  }
  __syncthreads();
#pragma unroll
  for (int tt = 0; tt < 2; ++tt) {
    v8f acc = tile_gemm<HID>(&Akj[0][0], HID, pWv, 2 * w + tt, lane);
    int col = w * HD + tt * 16 + (lane & 15);
    float bb = bv[col];
    int rbase = (lane >> 4) * 8;
#pragma unroll
    for (int j = 0; j < 8; ++j) {
      int row = rbase + j;
      atomicAdd(&agg[(size_t)dstS[row] * HID + col], (acc[j] + bb) * att[w][row]);
    }
  }
}

// h = LN(x + agg@Wo + bo)
__global__ __launch_bounds__(128) void k_node1(
    const float* __restrict__ x, const float* __restrict__ agg,
    const _Float16* __restrict__ pWo, const float* __restrict__ bo, float* __restrict__ h) {
  __shared__ _Float16 A[16][HID];
  __shared__ float Y[16][HID];
  __shared__ float red1[16][8], red2[16][8], mu[16], rs[16];
  const int n0b = blockIdx.x * 16;
  const int t = threadIdx.x, lane = t & 31, w = t >> 5;
  for (int r = 0; r < 16; ++r) A[r][t] = (_Float16)agg[(size_t)(n0b + r) * HID + t];
  __syncthreads();
#pragma unroll
  for (int tt = 0; tt < 2; ++tt) {
    v8f acc = tile_gemm<HID>(&A[0][0], HID, pWo, 2 * w + tt, lane);
    int col = w * 32 + tt * 16 + (lane & 15);
    float bb = bo[col];
    int rbase = (lane >> 4) * 8;
#pragma unroll
    for (int j = 0; j < 8; ++j) {
      int row = rbase + j;
      Y[row][col] = x[(size_t)(n0b + row) * HID + col] + acc[j] + bb;
    }
  }
  __syncthreads();
  ln_store(Y, red1, red2, mu, rs, t, h + (size_t)n0b * HID);
}

// edge conv: gelu([h_dst,h_src,ge]@Wm1+bm1)@Wm2+bm2, * (1+dw), scatter into agg2
__global__ __launch_bounds__(128) void k_conv(
    const float* __restrict__ h, const int* __restrict__ ei, const float* __restrict__ ge,
    const float* __restrict__ qd,
    const _Float16* __restrict__ pWm1, const float* __restrict__ bm1,
    const _Float16* __restrict__ pWm2, const float* __restrict__ bm2,
    const float* __restrict__ wd, const float* __restrict__ bd, float* __restrict__ agg2) {
  __shared__ _Float16 A[16][3 * HID];
  __shared__ _Float16 Mid[16][2 * HID];
  __shared__ float dwS[16];
  __shared__ int dstS[16];
  const int e0 = blockIdx.x * 16;
  const int t = threadIdx.x, lane = t & 31, w = t >> 5;
  const int* src = ei;
  const int* dst = ei + NEDGE;
  for (int r = 0; r < 16; ++r) {
    int e = e0 + r;
    int s = src[e], d = dst[e];
    A[r][t]           = (_Float16)h[(size_t)d * HID + t];
    A[r][HID + t]     = (_Float16)h[(size_t)s * HID + t];
    A[r][2 * HID + t] = (_Float16)ge[(size_t)e * HID + t];
  }
  if (t < 16) {
    int e = e0 + t;
    dstS[t] = dst[e];
    float q = qd[e], s = 0.f;
#pragma unroll
    for (int hh = 0; hh < NH; ++hh) s += sigm_f(q * wd[hh] + bd[hh]);
    dwS[t] = s * (1.f / NH);
  }
  if (e0 + 16 < NEDGE) __builtin_prefetch(ge + (size_t)(e0 + 16) * HID, 0, 1);
  __syncthreads();
  // GEMM1: [16x384]@[384x256] -> gelu -> Mid
#pragma unroll
  for (int tt = 0; tt < 4; ++tt) {
    v8f acc = tile_gemm<3 * HID>(&A[0][0], 3 * HID, pWm1, 4 * w + tt, lane);
    int col = w * 64 + tt * 16 + (lane & 15);
    float bb = bm1[col];
    int rbase = (lane >> 4) * 8;
#pragma unroll
    for (int j = 0; j < 8; ++j) Mid[rbase + j][col] = (_Float16)gelu_f(acc[j] + bb);
  }
  __syncthreads();
  // GEMM2: [16x256]@[256x128] -> scale -> scatter
#pragma unroll
  for (int tt = 0; tt < 2; ++tt) {
    v8f acc = tile_gemm<2 * HID>(&Mid[0][0], 2 * HID, pWm2, 2 * w + tt, lane);
    int col = w * 32 + tt * 16 + (lane & 15);
    float bb = bm2[col];
    int rbase = (lane >> 4) * 8;
#pragma unroll
    for (int j = 0; j < 8; ++j) {
      int row = rbase + j;
      atomicAdd(&agg2[(size_t)dstS[row] * HID + col], (acc[j] + bb) * (1.f + dwS[row]));
    }
  }
}

// h2 = LN(h*(1-g) + u*g), g=sigmoid(cat@Wg+bg), u=gelu(cat@Wu+bu), cat=[h,agg2]
__global__ __launch_bounds__(128) void k_node2(
    const float* __restrict__ h, const float* __restrict__ agg2,
    const _Float16* __restrict__ pWg, const float* __restrict__ bg,
    const _Float16* __restrict__ pWu, const float* __restrict__ bu, float* __restrict__ h2) {
  __shared__ _Float16 A[16][2 * HID];
  __shared__ float Y[16][HID];
  __shared__ float red1[16][8], red2[16][8], mu[16], rs[16];
  const int n0b = blockIdx.x * 16;
  const int t = threadIdx.x, lane = t & 31, w = t >> 5;
  for (int r = 0; r < 16; ++r) {
    A[r][t]       = (_Float16)h[(size_t)(n0b + r) * HID + t];
    A[r][HID + t] = (_Float16)agg2[(size_t)(n0b + r) * HID + t];
  }
  __syncthreads();
#pragma unroll
  for (int tt = 0; tt < 2; ++tt) {
    v8f ag = tile_gemm<2 * HID>(&A[0][0], 2 * HID, pWg, 2 * w + tt, lane);
    v8f au = tile_gemm<2 * HID>(&A[0][0], 2 * HID, pWu, 2 * w + tt, lane);
    int col = w * 32 + tt * 16 + (lane & 15);
    float bgv = bg[col], buv = bu[col];
    int rbase = (lane >> 4) * 8;
#pragma unroll
    for (int j = 0; j < 8; ++j) {
      int row = rbase + j;
      float g = sigm_f(ag[j] + bgv);
      float u = gelu_f(au[j] + buv);
      float hv = h[(size_t)(n0b + row) * HID + col];
      Y[row][col] = hv * (1.f - g) + u * g;
    }
  }
  __syncthreads();
  ln_store(Y, red1, red2, mu, rs, t, h2 + (size_t)n0b * HID);
}

// FF: hn=LN(h2); ff=gelu(hn@Wf1+bf1)@Wf2+bf2; gate=sigmoid([h2,ff]@Wgate+bgate);
// out = LN(h2 + gate*ff)
__global__ __launch_bounds__(128) void k_ff(
    const float* __restrict__ h2,
    const _Float16* __restrict__ pWf1, const float* __restrict__ bf1,
    const _Float16* __restrict__ pWf2, const float* __restrict__ bf2,
    const _Float16* __restrict__ pWgate, const float* __restrict__ bgate,
    float* __restrict__ out) {
  __shared__ _Float16 Ahn[16][HID];
  __shared__ _Float16 Ah2[16][HID];
  __shared__ _Float16 Mid[16][FFDIM];
  __shared__ _Float16 Aff[16][HID];
  __shared__ float Y[16][HID];
  __shared__ float red1[16][8], red2[16][8], mu[16], rs[16];
  const int n0b = blockIdx.x * 16;
  const int t = threadIdx.x, lane = t & 31, w = t >> 5;
  for (int r = 0; r < 16; ++r) Y[r][t] = h2[(size_t)(n0b + r) * HID + t];
  __syncthreads();
  {
    int r = t >> 3, seg = t & 7;
    float s1 = 0.f, s2 = 0.f;
#pragma unroll
    for (int i = 0; i < 16; ++i) { float v = Y[r][seg * 16 + i]; s1 += v; s2 += v * v; }
    red1[r][seg] = s1; red2[r][seg] = s2;
  }
  __syncthreads();
  if (t < 16) {
    float s1 = 0.f, s2 = 0.f;
#pragma unroll
    for (int i = 0; i < 8; ++i) { s1 += red1[t][i]; s2 += red2[t][i]; }
    float m = s1 * (1.f / HID), v = s2 * (1.f / HID) - m * m;
    mu[t] = m; rs[t] = rsqrtf(v + 1e-5f);
  }
  __syncthreads();
  for (int r = 0; r < 16; ++r) {
    Ahn[r][t] = (_Float16)((Y[r][t] - mu[r]) * rs[r]);
    Ah2[r][t] = (_Float16)Y[r][t];
  }
  __syncthreads();
  // GEMM1: hn @ Wf1 (128x512) -> gelu -> Mid; wave w owns cols [128w,128w+128)
#pragma unroll
  for (int tt = 0; tt < 8; ++tt) {
    v8f acc = tile_gemm<HID>(&Ahn[0][0], HID, pWf1, 8 * w + tt, lane);
    int col = w * 128 + tt * 16 + (lane & 15);
    float bb = bf1[col];
    int rbase = (lane >> 4) * 8;
#pragma unroll
    for (int j = 0; j < 8; ++j) Mid[rbase + j][col] = (_Float16)gelu_f(acc[j] + bb);
  }
  __syncthreads();
  // GEMM2: Mid @ Wf2 (512x128); keep ff in regs + f16 copy for gate GEMM
  v8f ff0 = {}, ff1 = {};
#pragma unroll
  for (int tt = 0; tt < 2; ++tt) {
    v8f acc = tile_gemm<FFDIM>(&Mid[0][0], FFDIM, pWf2, 2 * w + tt, lane);
    int col = w * 32 + tt * 16 + (lane & 15);
    float bb = bf2[col];
    int rbase = (lane >> 4) * 8;
#pragma unroll
    for (int j = 0; j < 8; ++j) {
      acc[j] += bb;
      Aff[rbase + j][col] = (_Float16)acc[j];
    }
    if (tt == 0) ff0 = acc; else ff1 = acc;
  }
  __syncthreads();
  // GEMM3: [h2, ff] @ Wgate (256x128) -> gate; combine into Y
#pragma unroll
  for (int tt = 0; tt < 2; ++tt) {
    int nt = 2 * w + tt;
    v8f acc = {};
#pragma unroll
    for (int kk = 0; kk < 8; ++kk) {
      v16h a = (kk < 4) ? ld_a(&Ah2[0][0], HID, kk * 32, lane)
                        : ld_a(&Aff[0][0], HID, kk * 32 - 128, lane);
      acc = wmma_step(a, ld_bp(pWgate, 8, kk, nt, lane), acc);
    }
    int col = w * 32 + tt * 16 + (lane & 15);
    float bb = bgate[col];
    int rbase = (lane >> 4) * 8;
    v8f ff = (tt == 0) ? ff0 : ff1;
#pragma unroll
    for (int j = 0; j < 8; ++j) {
      int row = rbase + j;
      float g = sigm_f(acc[j] + bb);
      Y[row][col] = Y[row][col] + g * ff[j];   // h2 + gate*ff (Y still holds h2)
    }
  }
  __syncthreads();
  ln_store(Y, red1, red2, mu, rs, t, out + (size_t)n0b * HID);
}

// ---------------- launcher ----------------
extern "C" void kernel_launch(void* const* d_in, const int* in_sizes, int n_in,
                              void* d_out, int out_size, void* d_ws, size_t ws_size,
                              hipStream_t stream) {
  (void)in_sizes; (void)n_in; (void)out_size; (void)ws_size;
  const float* x    = (const float*)d_in[0];
  const int*   ei   = (const int*)d_in[1];
  const float* ge   = (const float*)d_in[2];
  const float* tp   = (const float*)d_in[3];
  const float* qd   = (const float*)d_in[4];
  const float* Wq   = (const float*)d_in[5];  const float* bq   = (const float*)d_in[6];
  const float* Wk   = (const float*)d_in[7];  const float* bk   = (const float*)d_in[8];
  const float* Wv   = (const float*)d_in[9];  const float* bv   = (const float*)d_in[10];
  const float* wt   = (const float*)d_in[11]; const float* bt   = (const float*)d_in[12];
  const float* Wo   = (const float*)d_in[13]; const float* bo   = (const float*)d_in[14];
  const float* Wm1  = (const float*)d_in[15]; const float* bm1  = (const float*)d_in[16];
  const float* Wm2  = (const float*)d_in[17]; const float* bm2  = (const float*)d_in[18];
  const float* wd   = (const float*)d_in[19]; const float* bd   = (const float*)d_in[20];
  const float* Wg   = (const float*)d_in[21]; const float* bg   = (const float*)d_in[22];
  const float* Wu   = (const float*)d_in[23]; const float* bu   = (const float*)d_in[24];
  const float* Wf1  = (const float*)d_in[25]; const float* bf1  = (const float*)d_in[26];
  const float* Wf2  = (const float*)d_in[27]; const float* bf2  = (const float*)d_in[28];
  const float* Wgate= (const float*)d_in[29]; const float* bgate= (const float*)d_in[30];

  float* ws = (float*)d_ws;
  float* sc       = ws;             ws += (size_t)NEDGE * NH;
  unsigned* smaxu = (unsigned*)ws;  ws += (size_t)NNODE * NH;
  float* ssum     = ws;             ws += (size_t)NNODE * NH;
  float* agg      = ws;             ws += (size_t)NNODE * HID;
  float* h        = ws;             ws += (size_t)NNODE * HID;
  float* agg2     = ws;             ws += (size_t)NNODE * HID;
  float* h2       = ws;             ws += (size_t)NNODE * HID;

  // packed f16 weights (WMMA B-fragment layout), ~852 KB total -> L2 resident
  _Float16* hp = (_Float16*)ws;
  _Float16* pWq    = hp; hp += 128 * 128;
  _Float16* pWk    = hp; hp += 128 * 128;
  _Float16* pWv    = hp; hp += 128 * 128;
  _Float16* pWo    = hp; hp += 128 * 128;
  _Float16* pWm1   = hp; hp += 384 * 256;
  _Float16* pWm2   = hp; hp += 256 * 128;
  _Float16* pWg    = hp; hp += 256 * 128;
  _Float16* pWu    = hp; hp += 256 * 128;
  _Float16* pWf1   = hp; hp += 128 * 512;
  _Float16* pWf2   = hp; hp += 512 * 128;
  _Float16* pWgate = hp; hp += 256 * 128;

  const int eb = NEDGE / 16;   // 37500
  const int nb = NNODE / 16;   // 3125

  k_pack<<<(128 * 128 + 255) / 256, 256, 0, stream>>>(Wq, 128, 128, pWq);
  k_pack<<<(128 * 128 + 255) / 256, 256, 0, stream>>>(Wk, 128, 128, pWk);
  k_pack<<<(128 * 128 + 255) / 256, 256, 0, stream>>>(Wv, 128, 128, pWv);
  k_pack<<<(128 * 128 + 255) / 256, 256, 0, stream>>>(Wo, 128, 128, pWo);
  k_pack<<<(384 * 256 + 255) / 256, 256, 0, stream>>>(Wm1, 384, 256, pWm1);
  k_pack<<<(256 * 128 + 255) / 256, 256, 0, stream>>>(Wm2, 256, 128, pWm2);
  k_pack<<<(256 * 128 + 255) / 256, 256, 0, stream>>>(Wg, 256, 128, pWg);
  k_pack<<<(256 * 128 + 255) / 256, 256, 0, stream>>>(Wu, 256, 128, pWu);
  k_pack<<<(128 * 512 + 255) / 256, 256, 0, stream>>>(Wf1, 128, 512, pWf1);
  k_pack<<<(512 * 128 + 255) / 256, 256, 0, stream>>>(Wf2, 512, 128, pWf2);
  k_pack<<<(256 * 128 + 255) / 256, 256, 0, stream>>>(Wgate, 256, 128, pWgate);

  k_init<<<(NNODE * HID + 255) / 256, 256, 0, stream>>>(smaxu, ssum, agg, agg2);
  k_scores<<<eb, 128, 0, stream>>>(x, ei, ge, tp, pWq, bq, pWk, bk, wt, bt, sc);
  k_smax<<<(NEDGE * NH + 255) / 256, 256, 0, stream>>>(ei, sc, smaxu);
  k_expsum<<<(NEDGE * NH + 255) / 256, 256, 0, stream>>>(ei, sc, smaxu, ssum);
  k_aggv<<<eb, 128, 0, stream>>>(x, ei, ge, pWv, bv, sc, ssum, agg);
  k_node1<<<nb, 128, 0, stream>>>(x, agg, pWo, bo, h);
  k_conv<<<eb, 128, 0, stream>>>(h, ei, ge, qd, pWm1, bm1, pWm2, bm2, wd, bd, agg2);
  k_node2<<<nb, 128, 0, stream>>>(h, agg2, pWg, bg, pWu, bu, h2);
  k_ff<<<nb, 128, 0, stream>>>(h2, pWf1, bf1, pWf2, bf2, pWgate, bgate, (float*)d_out);
}
